// selfattention_75797582840746
// MI455X (gfx1250) — matrix operand order repaired and checked by reading, not compiled, not run on hardware
//
#include <hip/hip_runtime.h>
#include <hip/hip_bf16.h>

#define HEADS 8
#define KDIM  128
#define BB    4
#define TT    2048
#define MROWS (BB * TT)        // 8192
#define NQKV  (HEADS * KDIM)   // 1024

typedef __attribute__((ext_vector_type(16))) __bf16 v16bf;
typedef __attribute__((ext_vector_type(8)))  __bf16 v8bf;
typedef __attribute__((ext_vector_type(8)))  float  v8f;
typedef __attribute__((ext_vector_type(4)))  unsigned int u32x4;
typedef __attribute__((ext_vector_type(8)))  int i32x8;
typedef __attribute__((ext_vector_type(4)))  int i32x4;

#if __has_builtin(__builtin_amdgcn_tensor_load_to_lds)
#define USE_TDM 1
#else
#define USE_TDM 0
#endif

static __device__ __forceinline__ v8f wmma_bf16(v16bf a, v16bf b, v8f c) {
    return __builtin_amdgcn_wmma_f32_16x16x32_bf16(false, a, false, b, (short)0, c,
                                                   false, false);
}

// A fragment (16x32, row-major, stride ld): lane L holds row (L&15);
// elems 0..7 at k0+(L>>4)*8, elems 8..15 at +16.
static __device__ __forceinline__ v16bf load_afrag(const __bf16* base, int ld,
                                                   int row0, int k0, int lane) {
    int m = lane & 15, hi = lane >> 4;
    const __bf16* p = base + (size_t)(row0 + m) * ld + k0 + hi * 8;
    v16bf a;
    ((v8bf*)&a)[0] = *(const v8bf*)p;
    ((v8bf*)&a)[1] = *(const v8bf*)(p + 16);
    return a;
}

// B fragment (32x16) from N-major source: lane L holds n = n0+(L&15);
// elems 0..7 at k0+(L>>4)*8, elems 8..15 at +16.
static __device__ __forceinline__ v16bf load_bfrag(const __bf16* base, int ld,
                                                   int n0, int k0, int lane) {
    int n = lane & 15, hi = lane >> 4;
    const __bf16* p = base + (size_t)(n0 + n) * ld + k0 + hi * 8;
    v16bf b;
    ((v8bf*)&b)[0] = *(const v8bf*)p;
    ((v8bf*)&b)[1] = *(const v8bf*)(p + 16);
    return b;
}

#if USE_TDM
// 2-D tile DMA via Tensor Data Mover (6-arg clang-23 builtin:
// (uint32x4 g0, int32x8 g1, int32x4, int32x4, int32x8, i32 cpol)).
// D# packing per CDNA5 ISA 8.3/8.4. All dims in 4-byte dwords (data_size=2).
static __device__ __forceinline__ void tdm_load_2d(unsigned lds_off, const void* gaddr,
                                                   unsigned tile_d0_dw, unsigned tile_d1,
                                                   unsigned tensor_d0_dw, unsigned tensor_d1,
                                                   unsigned stride_dw) {
    unsigned long long ga = (unsigned long long)(uintptr_t)gaddr;
    u32x4 g0;
    g0[0] = 1u;                                            // count=1, user descriptor
    g0[1] = lds_off;                                       // lds_addr (bytes)
    g0[2] = (unsigned)ga;                                  // global_addr[31:0]
    g0[3] = ((unsigned)(ga >> 32) & 0x01ffffffu)           // global_addr[56:32]
            | 0x80000000u;                                 // type=2 ("image")
    i32x8 g1;
    g1[0] = (int)(2u << 16);                               // data_size=4B; no multicast
    g1[1] = (int)((tensor_d0_dw & 0xffffu) << 16);         // tensor_dim0[15:0]
    g1[2] = (int)(((tensor_d0_dw >> 16) & 0xffffu) |
                  ((tensor_d1 & 0xffffu) << 16));          // td0[31:16] | td1[15:0]
    g1[3] = (int)(((tensor_d1 >> 16) & 0xffffu) |
                  ((tile_d0_dw & 0xffffu) << 16));         // td1[31:16] | tile_dim0
    g1[4] = (int)(tile_d1 & 0xffffu);                      // tile_dim1, tile_dim2=0
    g1[5] = (int)stride_dw;                                // tensor_dim0_stride[31:0]
    g1[6] = 0;                                             // stride[47:32], dim1_stride lo
    g1[7] = 0;
    i32x4 gz4 = {0, 0, 0, 0};
    i32x8 gz8 = {0, 0, 0, 0, 0, 0, 0, 0};
    __builtin_amdgcn_tensor_load_to_lds(g0, g1, gz4, gz4, gz8, 0);
}
#endif

// ---------------- conversion / transpose kernels ----------------

__global__ void k_cvt_bf16(const float* __restrict__ in, __bf16* __restrict__ out, int n) {
    int i = blockIdx.x * blockDim.x + threadIdx.x;
    if (i < n) out[i] = (__bf16)in[i];
}

// src: [R][C] f32 row-major -> dst: [C][R] bf16 row-major
__global__ void k_transpose_bf16(const float* __restrict__ src, __bf16* __restrict__ dst,
                                 int R, int C) {
    int i = blockIdx.x * blockDim.x + threadIdx.x;
    if (i < R * C) {
        int c = i / R, r = i % R;
        dst[i] = (__bf16)src[(size_t)r * C + c];
    }
}

// ---------------- fused QKV projection ----------------
__global__ void __launch_bounds__(256) k_qkv_proj(const __bf16* __restrict__ xbf,
                                                  const __bf16* __restrict__ wqkv_t,
                                                  __bf16* __restrict__ Kb,
                                                  __bf16* __restrict__ Qb,
                                                  __bf16* __restrict__ Vt) {
    int lane = threadIdx.x & 31;
    int gw = blockIdx.x * 8 + (threadIdx.x >> 5);
    int ntile = gw & 63;
    int mtile = (gw >> 6) & 511;
    int mat = gw >> 15;               // 0=w_k 1=w_q 2=w_v
    if (mat >= 3) return;
    const __bf16* Wt = wqkv_t + (size_t)mat * NQKV * KDIM;

    v8f c = {};
#pragma unroll
    for (int kc = 0; kc < 4; ++kc) {
        v16bf a = load_afrag(xbf, KDIM, mtile * 16, kc * 32, lane);
        v16bf b = load_bfrag(Wt, KDIM, ntile * 16, kc * 32, lane);
        c = wmma_bf16(a, b, c);
    }
    int hi = lane >> 4, n = lane & 15;
    int ng = ntile * 16 + n;
    int h = ng >> 7, kd = ng & 127;
#pragma unroll
    for (int v = 0; v < 8; ++v) {
        int row = mtile * 16 + v + 8 * hi;
        int b_ = row >> 11, t = row & (TT - 1);
        size_t bh = (size_t)(b_ * HEADS + h);
        __bf16 val = (__bf16)c[v];
        if (mat == 0)      Kb[(bh * TT + t) * KDIM + kd] = val;
        else if (mat == 1) Qb[(bh * TT + t) * KDIM + kd] = val;
        else               Vt[(bh * KDIM + kd) * TT + t] = val;
    }
}

// ---------------- flash attention with TDM-staged K/V tiles ----------------
// Block = 8 waves, one (b,h), 128 q rows. K/V chunks (32 keys) double-buffered
// in LDS; DMA by wave 0 via tensor_load_to_lds, overlap with WMMA on current.
__global__ void __launch_bounds__(256) k_attention(const __bf16* __restrict__ Qb,
                                                   const __bf16* __restrict__ Kb,
                                                   const __bf16* __restrict__ Vt,
                                                   __bf16* __restrict__ O) {
    // LDS: K[2][32][128] (16KB) | V[2][128][32] (16KB) | P[8][16*40] (10KB)
    __shared__ __attribute__((aligned(128))) char smem[43008];
    __bf16* kbuf0 = (__bf16*)(smem);
    __bf16* kbuf1 = (__bf16*)(smem + 8192);
    __bf16* vbuf0 = (__bf16*)(smem + 16384);
    __bf16* vbuf1 = (__bf16*)(smem + 24576);
    __bf16* pw = (__bf16*)(smem + 32768) + (threadIdx.x >> 5) * (16 * 40);

    int lane = threadIdx.x & 31;
    int wid = threadIdx.x >> 5;
    int blk = blockIdx.x;                 // 4*8*16 = 512 blocks
    int qblk = blk & 15;
    int h = (blk >> 4) & 7;
    int b = blk >> 7;
    size_t bh = (size_t)(b * HEADS + h);
    const __bf16* Qbase = Qb + bh * TT * KDIM;
    const __bf16* Kbase = Kb + bh * TT * KDIM;
    const __bf16* Vbase = Vt + bh * KDIM * TT;    // [128][T]
    int row0 = (qblk * 8 + wid) * 16;
    int hi = lane >> 4, n = lane & 15;

    auto stage = [&](int buf, int s0) {
#if USE_TDM
        if (wid == 0) {
            // K chunk: rows s0..s0+31 of [T][128] bf16 -> 64dw x 32 rows, stride 64dw
            tdm_load_2d((unsigned)(buf * 8192), Kbase + (size_t)s0 * KDIM,
                        64u, 32u, 64u, (unsigned)TT, 64u);
            // V chunk: cols s0..s0+31 of [128][T] bf16 -> 16dw x 128 rows, stride T/2 dw
            tdm_load_2d((unsigned)(16384 + buf * 8192), Vbase + s0,
                        16u, 128u, (unsigned)(TT / 2), 128u, (unsigned)(TT / 2));
        }
#else
        __bf16* lk = buf ? kbuf1 : kbuf0;
        __bf16* lv = buf ? vbuf1 : vbuf0;
        for (int idx = threadIdx.x; idx < 512; idx += 256) {
            int r = idx >> 4, o = (idx & 15) * 8;
            *(v8bf*)(lk + r * KDIM + o) =
                *(const v8bf*)(Kbase + (size_t)(s0 + r) * KDIM + o);
        }
        for (int idx = threadIdx.x; idx < 512; idx += 256) {
            int r = idx >> 2, o = (idx & 3) * 8;
            *(v8bf*)(lv + r * 32 + o) =
                *(const v8bf*)(Vbase + (size_t)r * TT + s0 + o);
        }
#endif
    };
    auto stage_fence = [&]() {
#if USE_TDM
        if (wid == 0) {
#if __has_builtin(__builtin_amdgcn_s_wait_tensorcnt)
            __builtin_amdgcn_s_wait_tensorcnt(0);
#else
            asm volatile("s_wait_tensorcnt 0x0" ::: "memory");
#endif
        }
#endif
        __syncthreads();
    };

    v16bf qa[4];
#pragma unroll
    for (int kc = 0; kc < 4; ++kc) qa[kc] = load_afrag(Qbase, KDIM, row0, kc * 32, lane);

    v8f o[8];
    v8f vzero = {};
#pragma unroll
    for (int nt = 0; nt < 8; ++nt) o[nt] = vzero;
    float mrow[8], lrow[8];
#pragma unroll
    for (int v = 0; v < 8; ++v) { mrow[v] = -1e30f; lrow[v] = 0.0f; }

    stage(0, 0);
    stage_fence();

    int pb = 0;
    for (int s0 = 0; s0 < TT; s0 += 32, pb ^= 1) {
        if (s0 + 32 < TT) stage(pb ^ 1, s0 + 32);
        const __bf16* lk = pb ? kbuf1 : kbuf0;
        const __bf16* lv = pb ? vbuf1 : vbuf0;

        // S tile: 16 q x 32 keys, K-dim 128 (from LDS K tile)
        v8f sc0 = {}, sc1 = {};
#pragma unroll
        for (int kc = 0; kc < 4; ++kc) {
            v16bf kb0 = load_bfrag(lk, KDIM, 0, kc * 32, lane);
            v16bf kb1 = load_bfrag(lk, KDIM, 16, kc * 32, lane);
            sc0 = wmma_bf16(qa[kc], kb0, sc0);
            sc1 = wmma_bf16(qa[kc], kb1, sc1);
        }

        // online softmax (row = v + 8*hi; 16 lanes hold the 16 columns)
        float p0[8], p1[8];
#pragma unroll
        for (int v = 0; v < 8; ++v) {
            float cm = fmaxf(sc0[v], sc1[v]);
            cm = fmaxf(cm, __shfl_xor(cm, 1, 32));
            cm = fmaxf(cm, __shfl_xor(cm, 2, 32));
            cm = fmaxf(cm, __shfl_xor(cm, 4, 32));
            cm = fmaxf(cm, __shfl_xor(cm, 8, 32));
            float mnew = fmaxf(mrow[v], cm);
            float scale = __expf(mrow[v] - mnew);
            mrow[v] = mnew;
            lrow[v] *= scale;
#pragma unroll
            for (int nt = 0; nt < 8; ++nt) o[nt][v] *= scale;
            float e0 = __expf(sc0[v] - mnew);
            float e1 = __expf(sc1[v] - mnew);
            p0[v] = e0; p1[v] = e1;
            float rs = e0 + e1;
            rs += __shfl_xor(rs, 1, 32);
            rs += __shfl_xor(rs, 2, 32);
            rs += __shfl_xor(rs, 4, 32);
            rs += __shfl_xor(rs, 8, 32);
            lrow[v] += rs;
        }

        // C-layout -> A-layout via private LDS slice (in-order per wave)
#pragma unroll
        for (int v = 0; v < 8; ++v) {
            int m = v + 8 * hi;
            pw[m * 40 + n] = (__bf16)p0[v];
            pw[m * 40 + 16 + n] = (__bf16)p1[v];
        }
        v16bf pa;
        {
            const __bf16* pr = pw + (lane & 15) * 40 + hi * 8;
            ((v8bf*)&pa)[0] = *(const v8bf*)pr;
            ((v8bf*)&pa)[1] = *(const v8bf*)(pr + 16);
        }

        // O += P(16x32) @ V(32x128) from LDS V tile (N-major, stride 32)
#pragma unroll
        for (int nt = 0; nt < 8; ++nt) {
            v16bf vb = load_bfrag(lv, 32, nt * 16, 0, lane);
            o[nt] = wmma_bf16(pa, vb, o[nt]);
        }

        stage_fence();
    }

    float inv[8];
#pragma unroll
    for (int v = 0; v < 8; ++v) inv[v] = 1.0f / lrow[v];
#pragma unroll
    for (int nt = 0; nt < 8; ++nt) {
#pragma unroll
        for (int v = 0; v < 8; ++v) {
            int m = v + 8 * hi;
            size_t row = (size_t)b * TT + row0 + m;
            O[row * NQKV + h * KDIM + nt * 16 + n] = (__bf16)(o[nt][v] * inv[v]);
        }
    }
}

// ---------------- output projection: out = O(8192x1024) @ w_u(1024x128) ----------------
__global__ void __launch_bounds__(256) k_out_proj(const __bf16* __restrict__ O,
                                                  const __bf16* __restrict__ wu_t,
                                                  float* __restrict__ out) {
    int lane = threadIdx.x & 31;
    int gw = blockIdx.x * 8 + (threadIdx.x >> 5);
    int ntile = gw & 7;
    int mtile = gw >> 3;
    v8f c = {};
#pragma unroll 4
    for (int kc = 0; kc < 32; ++kc) {
        v16bf a = load_afrag(O, NQKV, mtile * 16, kc * 32, lane);
        v16bf b = load_bfrag(wu_t, NQKV, ntile * 16, kc * 32, lane);
        c = wmma_bf16(a, b, c);
    }
    int hi = lane >> 4, n = lane & 15;
#pragma unroll
    for (int v = 0; v < 8; ++v) {
        int row = mtile * 16 + v + 8 * hi;
        out[(size_t)row * KDIM + ntile * 16 + n] = c[v];
    }
}

// ---------------- launcher ----------------
extern "C" void kernel_launch(void* const* d_in, const int* in_sizes, int n_in,
                              void* d_out, int out_size, void* d_ws, size_t ws_size,
                              hipStream_t stream) {
    const float* x   = (const float*)d_in[0];
    const float* w_k = (const float*)d_in[1];
    const float* w_q = (const float*)d_in[2];
    const float* w_v = (const float*)d_in[3];
    const float* w_u = (const float*)d_in[4];
    float* out = (float*)d_out;

    char* ws = (char*)d_ws;
    __bf16* xbf   = (__bf16*)(ws);                                   // 2 MB
    __bf16* wqkvt = (__bf16*)(ws + (2u << 20));                      // 3 x 256 KB
    __bf16* wut   = (__bf16*)(ws + (2u << 20) + 786432u);            // 256 KB
    char* big = ws + 3145728u;
    __bf16* Qb = (__bf16*)(big);                                     // 16 MB
    __bf16* Kb = (__bf16*)(big + 16777216u);                         // 16 MB
    __bf16* Vt = (__bf16*)(big + 2u * 16777216u);                    // 16 MB
    __bf16* O  = (__bf16*)(big + 3u * 16777216u);                    // 16 MB

    k_cvt_bf16<<<(MROWS * KDIM + 255) / 256, 256, 0, stream>>>(x, xbf, MROWS * KDIM);
    k_transpose_bf16<<<(KDIM * NQKV + 255) / 256, 256, 0, stream>>>(w_k, wqkvt, KDIM, NQKV);
    k_transpose_bf16<<<(KDIM * NQKV + 255) / 256, 256, 0, stream>>>(w_q, wqkvt + NQKV * KDIM, KDIM, NQKV);
    k_transpose_bf16<<<(KDIM * NQKV + 255) / 256, 256, 0, stream>>>(w_v, wqkvt + 2 * NQKV * KDIM, KDIM, NQKV);
    k_transpose_bf16<<<(NQKV * KDIM + 255) / 256, 256, 0, stream>>>(w_u, wut, NQKV, KDIM);

    k_qkv_proj<<<12288, 256, 0, stream>>>(xbf, wqkvt, Kb, Qb, Vt);
    k_attention<<<512, 256, 0, stream>>>(Qb, Kb, Vt, O);
    k_out_proj<<<512, 256, 0, stream>>>(O, wut, out);
}
